// MapCountInfo_52939766891115
// MI455X (gfx1250) — matrix operand order repaired and checked by reading, not compiled, not run on hardware
//
#include <hip/hip_runtime.h>
#include <hip/hip_bf16.h>
#include <math.h>

typedef _Float16 v16h __attribute__((ext_vector_type(16)));
typedef _Float16 v8h  __attribute__((ext_vector_type(8)));
typedef _Float16 v4h  __attribute__((ext_vector_type(4)));
typedef float    v8f  __attribute__((ext_vector_type(8)));

#define BATCH 4096
#define LSEQ  64
#define LNWF  32

// ---- workspace layout (f16 buffers, base 256B aligned from hipMalloc) ----
#define WS_SEQW_OFF    0                         // [256 N][96 K] f16 col-major  (48 KB)
#define WS_NWFW_OFF    (WS_SEQW_OFF + 256*96*2)  // [6][32 N][32 K] f16          (12 KB)
#define WS_REDW_OFF    (WS_NWFW_OFF + 6*32*32*2) // [64 N][256 K] f16 col-major  (32 KB)
#define WS_FEATS_OFF   (WS_REDW_OFF + 64*256*2)  // [4096][256] f16              (2 MB)

static __device__ __forceinline__ v8f wmma16x16x32(v16h a, v16h b, v8f c) {
  return __builtin_amdgcn_wmma_f32_16x16x32_f16(false, a, false, b, (short)0, c, false, false);
}

// Build a v16h A/B operand from two contiguous 8-half runs.
static __device__ __forceinline__ v16h cat8(const _Float16* lo, const _Float16* hi) {
  v8h a = *(const v8h*)lo;
  v8h b = *(const v8h*)hi;
  return __builtin_shufflevector(a, b, 0,1,2,3,4,5,6,7,8,9,10,11,12,13,14,15);
}

// Branchless fast sigmoid/tanh on raw v_exp_f32 / v_rcp_f32 (keeps EXEC all-ones).
// Guard AMDGCN builtins on the DEVICE pass only; host pass gets libm fallbacks.
#if defined(__HIP_DEVICE_COMPILE__) && __has_builtin(__builtin_amdgcn_exp2f)
#define EXP2F(x) __builtin_amdgcn_exp2f(x)
#else
#define EXP2F(x) exp2f(x)
#endif
#if defined(__HIP_DEVICE_COMPILE__) && __has_builtin(__builtin_amdgcn_rcpf)
#define RCPF(x) __builtin_amdgcn_rcpf(x)
#else
#define RCPF(x) (1.0f / (x))
#endif
#define LOG2E 1.44269504088896f
static __device__ __forceinline__ float fsigm(float x) {
  return RCPF(1.0f + EXP2F(-LOG2E * x));
}
static __device__ __forceinline__ float ftanh(float x) {
  return 2.0f * RCPF(1.0f + EXP2F(-2.0f * LOG2E * x)) - 1.0f;
}

// ============================================================================
// K0a: build f16 weight buffers in workspace (augmented: [Wh; Wx; bias; 0]).
// ============================================================================
__global__ void prep_weights_kernel(const float* seq_Wx, const float* seq_Wh, const float* seq_b,
                                    const float* nwf_Wx, const float* nwf_Wh, const float* nwf_b,
                                    const float* reduce_W, char* ws) {
  _Float16* seqW = (_Float16*)(ws + WS_SEQW_OFF);
  _Float16* nwfW = (_Float16*)(ws + WS_NWFW_OFF);
  _Float16* redW = (_Float16*)(ws + WS_REDW_OFF);
  const int T1 = 256 * 96, T2 = 6 * 32 * 32, T3 = 64 * 256;
  for (int idx = blockIdx.x * blockDim.x + threadIdx.x; idx < T1 + T2 + T3;
       idx += gridDim.x * blockDim.x) {
    if (idx < T1) {
      int n = idx / 96, k = idx % 96;
      float v = 0.0f;
      if (k < 64)       v = seq_Wh[k * 256 + n];
      else if (k < 70)  v = seq_Wx[(k - 64) * 256 + n];
      else if (k == 70) v = seq_b[n];
      seqW[n * 96 + k] = (_Float16)v;
    } else if (idx < T1 + T2) {
      int r = idx - T1;
      int m = r / 1024; r %= 1024;
      int n = r / 32, k = r % 32;
      float v = 0.0f;
      if (k < 8)        v = nwf_Wh[(m * 8 + k) * 32 + n];
      else if (k < 16)  v = nwf_Wx[(m * 8 + (k - 8)) * 32 + n];
      else if (k == 16) v = nwf_b[m * 32 + n];
      nwfW[m * 1024 + n * 32 + k] = (_Float16)v;
    } else {
      int r = idx - T1 - T2;
      int n = r / 256, k = r % 256;
      float v = (k < 234) ? reduce_W[k * 64 + n] : 0.0f;
      redW[n * 256 + k] = (_Float16)v;
    }
  }
}

// ============================================================================
// K0b: small embedding features -> feats f16 [4096][256].
// ============================================================================
__global__ void prep_feats_kernel(const int* goby_idx, const int* is_indel, const int* matches_ref,
                                  const int* count_fwd, const int* count_rev,
                                  const float* goby_emb, const float* bool_emb, const float* count_emb,
                                  char* ws) {
  int b = blockIdx.x * blockDim.x + threadIdx.x;
  if (b >= BATCH) return;
  _Float16* fr = (_Float16*)(ws + WS_FEATS_OFF) + (size_t)b * 256;
  int g = goby_idx[b];
  for (int k = 0; k < 4; ++k) fr[k] = (_Float16)goby_emb[g * 4 + k];
  int ii = is_indel[b];
  fr[4] = (_Float16)bool_emb[ii * 2 + 0];
  fr[5] = (_Float16)bool_emb[ii * 2 + 1];
  int mr = matches_ref[b];
  fr[6] = (_Float16)bool_emb[mr * 2 + 0];
  fr[7] = (_Float16)bool_emb[mr * 2 + 1];
  int cf = count_fwd[b], cr = count_rev[b];
  for (int k = 0; k < 5; ++k) fr[136 + k] = (_Float16)count_emb[cf * 5 + k];
  for (int k = 0; k < 5; ++k) fr[141 + k] = (_Float16)count_emb[cr * 5 + k];
  for (int k = 234; k < 256; ++k) fr[k] = (_Float16)0.0f;
}

// ============================================================================
// K1: sequence LSTM (E=6, H=64, T=64), gates via WMMA on augmented input
//     a_t = [h(64), x(6), 1, 0-pad] (K=96).  grid = (256 tiles, 2 seqs), 256 thr.
//     Wave w owns gate columns [32w, 32w+32): 2 N-tiles x 3 K-chunks = 6 WMMA/step.
//     Sequence indices + base-embedding table preloaded to LDS: the recurrence
//     loop touches no global memory.
// ============================================================================
__global__ void __launch_bounds__(256)
seq_lstm_kernel(const int* from_seq, const int* to_seq, const float* base_emb, char* ws) {
  __shared__ _Float16 aBuf[16][104];   // [row][K] augmented input, padded stride
  __shared__ float    gBuf[16][260];   // [row][gate] f32 gates, padded stride
  __shared__ int      sIdx[16][64];    // per-row sequence indices
  __shared__ _Float16 embL[85][6];     // base embedding table (f16)

  const int tid  = threadIdx.x;
  const int w    = tid >> 5;
  const int lane = tid & 31;
  const int b0   = blockIdx.x * 16;
  const int* seq = (blockIdx.y == 0) ? from_seq : to_seq;
  const int colbase = 8 + 64 * blockIdx.y;   // feats column base (m_from=8, m_to=72)

  const _Float16* seqW  = (const _Float16*)(ws + WS_SEQW_OFF);
  _Float16*       feats = (_Float16*)(ws + WS_FEATS_OFF);

  // Preload B operands (stay in VGPRs across the whole time loop).
  // B layout: lane -> column N = lane&15; halves -> K = 32*kc + p + (lane>=16 ? 16 : 0)
  v16h Bop[2][3];
  {
    const int nlo = (lane & 15);
    const int khi = (lane >= 16) ? 16 : 0;
    for (int nt = 0; nt < 2; ++nt) {
      int col = 32 * w + 16 * nt + nlo;
      for (int kc = 0; kc < 3; ++kc)
        Bop[nt][kc] = *(const v16h*)(seqW + col * 96 + 32 * kc + khi);
    }
  }

  // Preload indices (coalesced) and embedding table into LDS.
  for (int idx = tid; idx < 16 * 64; idx += 256) {
    int r = idx >> 6, t = idx & 63;
    sIdx[r][t] = seq[(b0 + r) * LSEQ + t];
  }
  for (int idx = tid; idx < 85 * 6; idx += 256)
    ((_Float16*)embL)[idx] = (_Float16)base_emb[idx];
  // Init A buffer: h=0, col 70 = 1 (bias row), pad 0.
  for (int idx = tid; idx < 16 * 104; idx += 256) ((_Float16*)aBuf)[idx] = (_Float16)0.0f;
  __syncthreads();
  if (tid < 16) {
    aBuf[tid][70] = (_Float16)1.0f;
    int s = sIdx[tid][0];
    for (int e = 0; e < 6; ++e) aBuf[tid][64 + e] = embL[s][e];
  }

  float c[4] = {0.f, 0.f, 0.f, 0.f};
  const int row = tid >> 4;          // elementwise ownership: (row, j0..j0+3)
  const int j0  = (tid & 15) * 4;

  const int rowA  = lane & 15;       // A layout: lane -> M, two 8-half runs per chunk
  const int koffA = (lane >= 16) ? 16 : 0;
  const int colD  = 32 * w + (lane & 15);
  const int rbase = (lane >= 16) ? 8 : 0;

  for (int t = 0; t < LSEQ; ++t) {
    __syncthreads();                 // aBuf ready
    v16h A0 = cat8(&aBuf[rowA][ 0 + koffA], &aBuf[rowA][ 8 + koffA]);
    v16h A1 = cat8(&aBuf[rowA][32 + koffA], &aBuf[rowA][40 + koffA]);
    v16h A2 = cat8(&aBuf[rowA][64 + koffA], &aBuf[rowA][72 + koffA]);
    v8f acc0 = {}, acc1 = {};
    acc0 = wmma16x16x32(A0, Bop[0][0], acc0);
    acc1 = wmma16x16x32(A0, Bop[1][0], acc1);
    acc0 = wmma16x16x32(A1, Bop[0][1], acc0);
    acc1 = wmma16x16x32(A1, Bop[1][1], acc1);
    acc0 = wmma16x16x32(A2, Bop[0][2], acc0);
    acc1 = wmma16x16x32(A2, Bop[1][2], acc1);
    // D layout: lane -> N = lane&15; reg r -> M = r + (lane>=16 ? 8 : 0)
    for (int r = 0; r < 8; ++r) {
      gBuf[rbase + r][colD]      = acc0[r];
      gBuf[rbase + r][colD + 16] = acc1[r];
    }
    __syncthreads();                 // gates ready
    // LSTM cell (gate order i,f,g,o at cols j, j+64, j+128, j+192)
    v4h hh;
    for (int q = 0; q < 4; ++q) {
      int j = j0 + q;
      float gi = gBuf[row][j], gf = gBuf[row][j + 64];
      float gg = gBuf[row][j + 128], go = gBuf[row][j + 192];
      c[q] = fsigm(gf) * c[q] + fsigm(gi) * ftanh(gg);
      hh[q] = (_Float16)(fsigm(go) * ftanh(c[q]));
    }
    *(v4h*)&aBuf[row][j0] = hh;      // h for next step
    if (t < LSEQ - 1 && tid < 16) {  // x_{t+1} from LDS-resident tables
      int s = sIdx[tid][t + 1];
      for (int e = 0; e < 6; ++e) aBuf[tid][64 + e] = embL[s][e];
    }
  }
  __syncthreads();
  // Final h (f16, already in aBuf cols 0..63) -> feature buffer.
  for (int idx = tid; idx < 16 * 64; idx += 256) {
    int r = idx >> 6, j = idx & 63;
    feats[(size_t)(b0 + r) * 256 + colbase + j] = aBuf[r][j];
  }
}

// ============================================================================
// K2: NWF LSTM (E=8, H=8, T=32).  a_t = [h(8), ne(4), fe(4), 1, pad] (K=32, one chunk).
//     grid = (256 tiles, 11 fields), 64 threads (2 waves, 1 WMMA each per step).
// ============================================================================
__global__ void __launch_bounds__(64)
nwf_lstm_kernel(const int* nwf_numbers, const int* nwf_freqs,
                const float* nwf_num_emb, const float* nwf_freq_emb, char* ws) {
  __shared__ _Float16 aBuf[16][40];
  __shared__ float    gBuf[16][36];
  __shared__ int      nIdx[16][32];
  __shared__ int      fIdx[16][32];

  const int tid  = threadIdx.x;
  const int w    = tid >> 5;
  const int lane = tid & 31;
  const int b0    = blockIdx.x * 16;
  const int field = blockIdx.y;
  const int FM[11] = {0, 0, 1, 1, 2, 2, 3, 3, 4, 5, 5};
  const int m = FM[field];

  const _Float16* nwfW  = (const _Float16*)(ws + WS_NWFW_OFF) + m * 1024;
  _Float16*       feats = (_Float16*)(ws + WS_FEATS_OFF);
  const int* nums  = nwf_numbers + (size_t)field * BATCH * LNWF;
  const int* freqs = nwf_freqs   + (size_t)field * BATCH * LNWF;
  const float* ne = nwf_num_emb  + (size_t)m * 1000 * 4;
  const float* fe = nwf_freq_emb + (size_t)m * 1000 * 4;

  v16h Bop = *(const v16h*)(nwfW + (16 * w + (lane & 15)) * 32 + ((lane >= 16) ? 16 : 0));

  // Preload index arrays (coalesced).
  for (int idx = tid; idx < 16 * 32; idx += 64) {
    int r = idx >> 5, t = idx & 31;
    nIdx[r][t] = nums[(b0 + r) * LNWF + t];
    fIdx[r][t] = freqs[(b0 + r) * LNWF + t];
  }
  for (int idx = tid; idx < 16 * 40; idx += 64) ((_Float16*)aBuf)[idx] = (_Float16)0.0f;
  __syncthreads();
  if (tid < 16) {
    aBuf[tid][16] = (_Float16)1.0f;  // bias row
    int n = nIdx[tid][0], f = fIdx[tid][0];
    for (int e = 0; e < 4; ++e) {
      aBuf[tid][8 + e]  = (_Float16)ne[n * 4 + e];
      aBuf[tid][12 + e] = (_Float16)fe[f * 4 + e];
    }
  }

  float c2[2] = {0.f, 0.f};
  const int row = tid >> 2;          // 16 rows x 8 units / 64 threads = 2 each
  const int j0  = (tid & 3) * 2;
  const int rowA  = lane & 15;
  const int koffA = (lane >= 16) ? 16 : 0;
  const int colD  = 16 * w + (lane & 15);
  const int rbase = (lane >= 16) ? 8 : 0;

  for (int t = 0; t < LNWF; ++t) {
    __syncthreads();
    v16h A = cat8(&aBuf[rowA][koffA], &aBuf[rowA][8 + koffA]);
    v8f acc = {};
    acc = wmma16x16x32(A, Bop, acc);
    for (int r = 0; r < 8; ++r) gBuf[rbase + r][colD] = acc[r];
    __syncthreads();
    for (int q = 0; q < 2; ++q) {
      int j = j0 + q;
      float gi = gBuf[row][j], gf = gBuf[row][j + 8];
      float gg = gBuf[row][j + 16], go = gBuf[row][j + 24];
      c2[q] = fsigm(gf) * c2[q] + fsigm(gi) * ftanh(gg);
      aBuf[row][j] = (_Float16)(fsigm(go) * ftanh(c2[q]));
    }
    if (t < LNWF - 1 && tid < 16) {
      int n = nIdx[tid][t + 1], f = fIdx[tid][t + 1];
      for (int e = 0; e < 4; ++e) {
        aBuf[tid][8 + e]  = (_Float16)ne[n * 4 + e];
        aBuf[tid][12 + e] = (_Float16)fe[f * 4 + e];
      }
    }
  }
  __syncthreads();
  for (int idx = tid; idx < 16 * 8; idx += 64) {
    int r = idx >> 3, u = idx & 7;
    feats[(size_t)(b0 + r) * 256 + 146 + 8 * field + u] = aBuf[r][u];
  }
}

// ============================================================================
// K3: final dense: relu(feats[4096,256]f16 @ redWT + bias) -> d_out f32 [4096,64].
// ============================================================================
__global__ void __launch_bounds__(128)
reduce_kernel(const float* reduce_b, char* ws, float* out) {
  const int tid  = threadIdx.x;
  const int w    = tid >> 5;
  const int lane = tid & 31;
  const int b0   = blockIdx.x * 16;

  const _Float16* feats = (const _Float16*)(ws + WS_FEATS_OFF);
  const _Float16* redW  = (const _Float16*)(ws + WS_REDW_OFF);

  const int rowA  = b0 + (lane & 15);
  const int koffA = (lane >= 16) ? 16 : 0;
  const int col   = 16 * w + (lane & 15);
  const int khiB  = (lane >= 16) ? 16 : 0;

  v8f acc = {};
  for (int kc = 0; kc < 8; ++kc) {
    const _Float16* ap = feats + (size_t)rowA * 256 + 32 * kc + koffA;
    v16h A = cat8(ap, ap + 8);
    v16h Bm = *(const v16h*)(redW + col * 256 + 32 * kc + khiB);
    acc = wmma16x16x32(A, Bm, acc);
  }
  float bias = reduce_b[col];
  const int rbase = (lane >= 16) ? 8 : 0;
  for (int r = 0; r < 8; ++r) {
    float v = acc[r] + bias;
    out[(size_t)(b0 + rbase + r) * 64 + col] = fmaxf(v, 0.0f);
  }
}

// ============================================================================
extern "C" void kernel_launch(void* const* d_in, const int* in_sizes, int n_in,
                              void* d_out, int out_size, void* d_ws, size_t ws_size,
                              hipStream_t stream) {
  const int* goby_idx    = (const int*)d_in[0];
  const int* is_indel    = (const int*)d_in[1];
  const int* matches_ref = (const int*)d_in[2];
  const int* from_seq    = (const int*)d_in[3];
  const int* to_seq      = (const int*)d_in[4];
  const int* count_fwd   = (const int*)d_in[5];
  const int* count_rev   = (const int*)d_in[6];
  const int* nwf_numbers = (const int*)d_in[7];
  const int* nwf_freqs   = (const int*)d_in[8];
  const float* goby_emb     = (const float*)d_in[9];
  const float* bool_emb     = (const float*)d_in[10];
  const float* count_emb    = (const float*)d_in[11];
  const float* base_emb     = (const float*)d_in[12];
  const float* seq_Wx       = (const float*)d_in[13];
  const float* seq_Wh       = (const float*)d_in[14];
  const float* seq_b        = (const float*)d_in[15];
  const float* nwf_num_emb  = (const float*)d_in[16];
  const float* nwf_freq_emb = (const float*)d_in[17];
  const float* nwf_Wx       = (const float*)d_in[18];
  const float* nwf_Wh       = (const float*)d_in[19];
  const float* nwf_b        = (const float*)d_in[20];
  const float* reduce_W     = (const float*)d_in[21];
  const float* reduce_b     = (const float*)d_in[22];
  char* ws = (char*)d_ws;
  float* out = (float*)d_out;

  prep_weights_kernel<<<64, 256, 0, stream>>>(seq_Wx, seq_Wh, seq_b,
                                              nwf_Wx, nwf_Wh, nwf_b, reduce_W, ws);
  prep_feats_kernel<<<BATCH / 256, 256, 0, stream>>>(goby_idx, is_indel, matches_ref,
                                                     count_fwd, count_rev,
                                                     goby_emb, bool_emb, count_emb, ws);
  seq_lstm_kernel<<<dim3(BATCH / 16, 2), 256, 0, stream>>>(from_seq, to_seq, base_emb, ws);
  nwf_lstm_kernel<<<dim3(BATCH / 16, 11), 64, 0, stream>>>(nwf_numbers, nwf_freqs,
                                                           nwf_num_emb, nwf_freq_emb, ws);
  reduce_kernel<<<BATCH / 16, 128, 0, stream>>>(reduce_b, ws, out);
}
